// ODE_RNN_39032662786483
// MI455X (gfx1250) — matrix operand order repaired and checked by reading, not compiled, not run on hardware
//
#include <hip/hip_runtime.h>
#include <hip/hip_bf16.h>

typedef _Float16 v16h __attribute__((ext_vector_type(16)));
typedef _Float16 v8h  __attribute__((ext_vector_type(8)));
typedef _Float16 v4h  __attribute__((ext_vector_type(4)));
typedef float    v8f  __attribute__((ext_vector_type(8)));
typedef float    v4f  __attribute__((ext_vector_type(4)));

namespace {
constexpr int Bsz = 256, Lseq = 2048, IN = 64, H = 128, HH = 128, OUTD = 64, G3 = 384;
constexpr int ROWS = 16;              // batch rows (M tile) per workgroup
constexpr int NWG  = Bsz / ROWS;      // 16 workgroups, one per WGP

// ---- LDS layout (bytes); weights f16 + f16 activation staging only ----
constexpr size_t OFF_W1   = 0;                                  // 128x128 f16
constexpr size_t OFF_W2   = OFF_W1   + (size_t)HH * H  * 2;     // 128x128 f16
constexpr size_t OFF_WHH  = OFF_W2   + (size_t)H  * HH * 2;     // 384x128 f16
constexpr size_t OFF_WIH  = OFF_WHH  + (size_t)G3 * H  * 2;     // 384x64  f16
constexpr size_t OFF_WOUT = OFF_WIH  + (size_t)G3 * IN * 2;     // 64x128  f16
constexpr size_t OFF_B1   = OFF_WOUT + (size_t)OUTD * H * 2;
constexpr size_t OFF_B2   = OFF_B1   + HH * 4;
constexpr size_t OFF_BIH  = OFF_B2   + H  * 4;
constexpr size_t OFF_BHH  = OFF_BIH  + G3 * 4;
constexpr size_t OFF_BOUT = OFF_BHH  + G3 * 4;
constexpr size_t OFF_H16A = OFF_BOUT + OUTD * 4;                // 16x128 f16 state (ping)
constexpr size_t OFF_H16B = OFF_H16A + ROWS * H * 2;            // 16x128 f16 state (pong)
constexpr size_t OFF_HA16 = OFF_H16B + ROWS * H * 2;            // 16x128 f16 rk4 stage input
constexpr size_t OFF_G16  = OFF_HA16 + ROWS * H * 2;            // 16x128 f16 relu intermediate
constexpr size_t OFF_X16  = OFF_G16  + ROWS * H * 2;            // 16x64  f16 x_t
constexpr size_t LDS_BYTES = OFF_X16 + ROWS * IN * 2;           // ~252 KB (< 320 KB/WGP)
}

// A-matrix 16x32 f16 fragment from LDS (row-major, `stride` halves per row).
// Layout used here (contiguous k, same as B): lane M = lane&15; halves 0..15 -> K = klo..klo+15,
// klo = (lane>>4)*16.
__device__ __forceinline__ v16h load_a_frag(const _Float16* base, int stride, int k0, int lane) {
  const int m   = lane & 15;
  const int klo = (lane >> 4) << 4;
  const _Float16* p = base + (size_t)m * stride + k0 + klo;
  v8h lo = *(const v8h*)(p);
  v8h hi = *(const v8h*)(p + 8);
  return __builtin_shufflevector(lo, hi, 0,1,2,3,4,5,6,7,8,9,10,11,12,13,14,15);
}

// B-matrix 32x16 f16 fragment: row-major weight [N][K] row is B^T; lane holds
// 16 contiguous halves K = (lane>>4)*16 + 0..15 of row N = tile*16 + (lane&15).
__device__ __forceinline__ v16h load_b_frag(const _Float16* brow_k0, int lane) {
  const _Float16* p = brow_k0 + ((lane >> 4) << 4);
  v8h lo = *(const v8h*)(p);
  v8h hi = *(const v8h*)(p + 8);
  return __builtin_shufflevector(lo, hi, 0,1,2,3,4,5,6,7,8,9,10,11,12,13,14,15);
}

__device__ __forceinline__ v8f splat8(float v) {
  v8f c;
#pragma unroll
  for (int i = 0; i < 8; ++i) c[i] = v;
  return c;
}

// GEMM 16x16 tile, register-resident B fragments.
template <int NK>
__device__ __forceinline__ v8f gemm_regB(const _Float16* A, int astride,
                                         const v16h* bf, v8f c, int lane) {
  v16h a[NK];
#pragma unroll
  for (int kt = 0; kt < NK; ++kt) a[kt] = load_a_frag(A, astride, kt * 32, lane);
#pragma unroll
  for (int kt = 0; kt < NK; ++kt)
    c = __builtin_amdgcn_wmma_f32_16x16x32_f16(false, a[kt], false, bf[kt], (short)0, c, false, false);
  return c;
}

// Gate nonlinearities via TRANS ops (v_exp + v_rcp); co-execute with WMMA.
__device__ __forceinline__ float dsigmoid(float x) {
  return __builtin_amdgcn_rcpf(1.0f + __expf(-x));
}
__device__ __forceinline__ float dtanh(float x) {
  float e2 = __expf(-2.0f * x);
  return (1.0f - e2) * __builtin_amdgcn_rcpf(1.0f + e2);
}
// ReLU as a single v_med3_f32: med(x, 0, +inf) == clamp(x, 0, inf)
__device__ __forceinline__ float drelu(float x) {
  return __builtin_amdgcn_fmed3f(x, 0.0f, __builtin_inff());
}

__global__ __launch_bounds__(256, 1)
void ode_rnn_wmma(const float* __restrict__ x,
                  const float* __restrict__ W1,  const float* __restrict__ b1,
                  const float* __restrict__ W2,  const float* __restrict__ b2,
                  const float* __restrict__ Wih, const float* __restrict__ bih,
                  const float* __restrict__ Whh, const float* __restrict__ bhh,
                  const float* __restrict__ Wout,const float* __restrict__ bout,
                  float* __restrict__ out) {
  extern __shared__ char smem[];
  _Float16* sW1   = (_Float16*)(smem + OFF_W1);
  _Float16* sW2   = (_Float16*)(smem + OFF_W2);
  _Float16* sWhh  = (_Float16*)(smem + OFF_WHH);
  _Float16* sWih  = (_Float16*)(smem + OFF_WIH);
  _Float16* sWout = (_Float16*)(smem + OFF_WOUT);
  float*    sb1   = (float*)(smem + OFF_B1);
  float*    sb2   = (float*)(smem + OFF_B2);
  float*    sbih  = (float*)(smem + OFF_BIH);
  float*    sbhh  = (float*)(smem + OFF_BHH);
  float*    sbout = (float*)(smem + OFF_BOUT);
  _Float16* h16a  = (_Float16*)(smem + OFF_H16A);
  _Float16* h16b  = (_Float16*)(smem + OFF_H16B);
  _Float16* ha16  = (_Float16*)(smem + OFF_HA16);
  _Float16* g16   = (_Float16*)(smem + OFF_G16);
  _Float16* sx16  = (_Float16*)(smem + OFF_X16);

  const int tid  = threadIdx.x;
  const int lane = tid & 31;
  const int wave = tid >> 5;                    // 8 waves, wave owns N-tile(s)
  const int wg   = blockIdx.x;                  // batch rows [wg*16, wg*16+16)
  const int nn   = lane & 15;                   // column within a 16-wide tile
  const int mb   = (lane >> 4) << 3;            // D-matrix row base for this lane
  const int ncol = wave * 16 + nn;              // column in H-wide buffers

  // ---- one-time: convert weights to f16 LDS, copy biases, zero h0 ----
  for (int i = tid; i < HH * H;  i += 256) sW1[i]  = (_Float16)W1[i];
  for (int i = tid; i < H * HH;  i += 256) sW2[i]  = (_Float16)W2[i];
  for (int i = tid; i < G3 * H;  i += 256) sWhh[i] = (_Float16)Whh[i];
  for (int i = tid; i < G3 * IN; i += 256) sWih[i] = (_Float16)Wih[i];
  for (int i = tid; i < OUTD * H;i += 256) sWout[i]= (_Float16)Wout[i];
  for (int i = tid; i < H;  i += 256) { sb1[i] = b1[i]; sb2[i] = b2[i]; }
  for (int i = tid; i < G3; i += 256) { sbih[i] = bih[i]; sbhh[i] = bhh[i]; }
  if (tid < OUTD) sbout[tid] = bout[tid];
  for (int i = tid; i < ROWS * H; i += 256) h16a[i] = (_Float16)0.0f;
  __syncthreads();

  // ---- hoist loop-invariant B fragments into VGPRs ----
  v16h w1f[4], w2f[4], whhf[3][4], wihf[3][2];
  float bihs[3], bhhs[3];
#pragma unroll
  for (int kt = 0; kt < 4; ++kt) w1f[kt] = load_b_frag(sW1 + (size_t)ncol * H + kt * 32, lane);
#pragma unroll
  for (int kt = 0; kt < 4; ++kt) w2f[kt] = load_b_frag(sW2 + (size_t)ncol * H + kt * 32, lane);
#pragma unroll
  for (int p = 0; p < 3; ++p) {
    int nc = (wave + p * 8) * 16 + nn;          // 0..383
    bihs[p] = sbih[nc];
    bhhs[p] = sbhh[nc];
#pragma unroll
    for (int kt = 0; kt < 4; ++kt) whhf[p][kt] = load_b_frag(sWhh + (size_t)nc * H + kt * 32, lane);
#pragma unroll
    for (int kt = 0; kt < 2; ++kt) wihf[p][kt] = load_b_frag(sWih + (size_t)nc * IN + kt * 32, lane);
  }
  const float b1s = sb1[ncol];
  const float b2s = sb2[ncol];

  // ---- register-resident state in the WMMA D layout: elem (mb+i, ncol) ----
  v8f hreg = splat8(0.0f);
  v8f S1r  = splat8(0.0f);
  v8f S2r  = splat8(0.0f);
  v8f Accr = splat8(0.0f);

  const float* xrow = x + ((size_t)(wg * ROWS + (tid >> 4)) * Lseq) * IN + (tid & 15) * 4;
  _Float16* sxdst = sx16 + (tid >> 4) * IN + (tid & 15) * 4;

#pragma unroll 1
  for (int t = 0; t < Lseq; ++t) {
    // ping-pong hidden-state buffers: removes the WAR barrier around the gates
    _Float16* hcur = (t & 1) ? h16b : h16a;     // state entering this step
    _Float16* hnxt = (t & 1) ? h16a : h16b;     // state produced by the GRU

    // ---- stage x_t [16,64] -> f16 LDS (vectorized); prefetch x_{t+1} ----
    {
      v4f xv = *(const v4f*)(xrow + (size_t)t * IN);
      v4h xh;
#pragma unroll
      for (int i = 0; i < 4; ++i) xh[i] = (_Float16)xv[i];
      *(v4h*)sxdst = xh;
      if (t + 1 < Lseq) __builtin_prefetch(xrow + (size_t)(t + 1) * IN, 0, 3);
    }

    // ================= RK4 (3/8 rule), 4 f-evals, state in VGPRs =================
#pragma unroll
    for (int s = 0; s < 4; ++s) {
      const _Float16* Ain = (s == 0) ? hcur : ha16;
      // g = relu(Ain @ W1^T + b1) -> g16
      {
        v8f c = gemm_regB<4>(Ain, H, w1f, splat8(b1s), lane);
#pragma unroll
        for (int r = 0; r < 8; ++r)
          g16[(mb + r) * H + ncol] = (_Float16)drelu(c[r]);
      }
      __syncthreads();
      // k = g @ W2^T + b2  (registers) ; fuse RK4 combination
      {
        v8f k = gemm_regB<4>(g16, H, w2f, splat8(b2s), lane);
        if (s == 0) {
          S1r = k; Accr = k;
          v8f ha = hreg + k * (1.0f / 3.0f);
#pragma unroll
          for (int r = 0; r < 8; ++r) ha16[(mb + r) * H + ncol] = (_Float16)ha[r];
        } else if (s == 1) {
          S2r = k; Accr += k * 3.0f;
          v8f ha = hreg + (k - S1r * (1.0f / 3.0f));
#pragma unroll
          for (int r = 0; r < 8; ++r) ha16[(mb + r) * H + ncol] = (_Float16)ha[r];
        } else if (s == 2) {
          Accr += k * 3.0f;
          v8f ha = hreg + (S1r - S2r + k);
#pragma unroll
          for (int r = 0; r < 8; ++r) ha16[(mb + r) * H + ncol] = (_Float16)ha[r];
        } else {
          hreg = hreg + (Accr + k) * 0.125f;    // h after rk4
#pragma unroll
          for (int r = 0; r < 8; ++r) hcur[(mb + r) * H + ncol] = (_Float16)hreg[r];
        }
      }
      __syncthreads();
    }

    // ================= GRU cell, gates fully in VGPRs =================
    // gx = x_t @ Wih^T + bih : this thread's three gate tiles share (m, ncol)
    v8f gxr = gemm_regB<2>(sx16, IN, wihf[0], splat8(bihs[0]), lane);
    v8f gxz = gemm_regB<2>(sx16, IN, wihf[1], splat8(bihs[1]), lane);
    v8f gxn = gemm_regB<2>(sx16, IN, wihf[2], splat8(bihs[2]), lane);
    // gh = h @ Whh^T (+ staged gx for r,z ; + bhh)
    v8f cr = gemm_regB<4>(hcur, H, whhf[0], gxr + splat8(bhhs[0]), lane);
    v8f cz = gemm_regB<4>(hcur, H, whhf[1], gxz + splat8(bhhs[1]), lane);
    v8f cn = gemm_regB<4>(hcur, H, whhf[2], splat8(bhhs[2]), lane);
    // gates in registers; results go to the *other* state buffer (no WAR barrier)
#pragma unroll
    for (int i = 0; i < 8; ++i) {
      float rg = dsigmoid(cr[i]);
      float zg = dsigmoid(cz[i]);
      float ng = dtanh(gxn[i] + rg * cn[i]);   // n = tanh(xn + r*hn)
      float hv = (1.0f - zg) * ng + zg * hreg[i];
      hreg[i] = hv;
      hnxt[(mb + i) * H + ncol] = (_Float16)hv;
    }
    __syncthreads();
  }

  // ================= final linear: out = h_last @ Wout^T + bout =================
  // After Lseq (even) steps the live state buffer is h16a.
  const _Float16* hfin = (Lseq & 1) ? h16b : h16a;
  if (wave < 4) {                               // OUT=64 -> 4 N-tiles
    v16h wof[4];
#pragma unroll
    for (int kt = 0; kt < 4; ++kt) wof[kt] = load_b_frag(sWout + (size_t)ncol * H + kt * 32, lane);
    v8f c = gemm_regB<4>(hfin, H, wof, splat8(sbout[ncol]), lane);
#pragma unroll
    for (int r = 0; r < 8; ++r)
      out[(size_t)(wg * ROWS + mb + r) * OUTD + ncol] = c[r];
  }
}

extern "C" void kernel_launch(void* const* d_in, const int* in_sizes, int n_in,
                              void* d_out, int out_size, void* d_ws, size_t ws_size,
                              hipStream_t stream) {
  const float* x    = (const float*)d_in[0];
  const float* W1   = (const float*)d_in[1];
  const float* b1   = (const float*)d_in[2];
  const float* W2   = (const float*)d_in[3];
  const float* b2   = (const float*)d_in[4];
  const float* Wih  = (const float*)d_in[5];
  const float* bih  = (const float*)d_in[6];
  const float* Whh  = (const float*)d_in[7];
  const float* bhh  = (const float*)d_in[8];
  const float* Wout = (const float*)d_in[9];
  const float* bout = (const float*)d_in[10];
  float* out = (float*)d_out;
  (void)in_sizes; (void)n_in; (void)out_size; (void)d_ws; (void)ws_size;
  hipLaunchKernelGGL(ode_rnn_wmma, dim3(NWG), dim3(256), LDS_BYTES, stream,
                     x, W1, b1, W2, b2, Wih, bih, Whh, bhh, Wout, bout, out);
}
